// PointFEF_16922171146620
// MI455X (gfx1250) — compile-verified, matrix operand order checked by reading, not run on hardware
//
#include <hip/hip_runtime.h>
#include <hip/hip_bf16.h>

typedef __bf16 bf16_t;
typedef __attribute__((ext_vector_type(16))) __bf16 v16bf;
typedef __attribute__((ext_vector_type(8)))  __bf16 v8bf;
typedef __attribute__((ext_vector_type(8)))  float   v8f;

#define EPSBN 1e-5f
#define BB 2
#define CU 3
#define CV 64
#define NN 4096
#define SS 64
#define CH 134           // 2*CU + 2*CV
#define KP 160           // CH padded to multiple of 32
#define CO 64            // out channels of Wuv / Wf
#define YSTR 72          // padded row stride of ytile
#define CNT1 ((float)(BB*NN*SS))
#define CNT2 ((float)(BB*NN))

// Async global->LDS b128 copy (gfx1250 GLOBAL_LOAD_ASYNC_TO_LDS_B128, ASYNCcnt).
__device__ __forceinline__ void async_copy_b128(unsigned lds_off, unsigned byte_off,
                                                const void* base) {
  asm volatile("global_load_async_to_lds_b128 %0, %1, %2"
               :: "v"(lds_off), "v"(byte_off), "s"(base) : "memory");
}
__device__ __forceinline__ void async_wait() {
  asm volatile("s_wait_asynccnt 0" ::: "memory");
}

// ---------------- prep: separable A (n-part) and B (s-part) terms ----------------
__global__ __launch_bounds__(256) void k_prepA(const float* __restrict__ u,
                                               const float* __restrict__ v,
                                               const float* __restrict__ Wgu,
                                               const float* __restrict__ Wgv,
                                               float* __restrict__ Aall) {
  int idx = blockIdx.x * 256 + threadIdx.x;
  if (idx >= BB * CH * NN) return;
  int n = idx % NN;
  int c = (idx / NN) % CH;
  int b = idx / (CH * NN);
  float acc = 0.f;
  if (c < 2 * CU) {
    for (int j = 0; j < CU; ++j)
      acc += Wgu[c * (2 * CU) + j] * u[((size_t)b * CU + j) * NN + n];
  } else {
    int o = c - 2 * CU;
    for (int j = 0; j < CV; ++j)
      acc += Wgv[o * (2 * CV) + j] * v[((size_t)b * CV + j) * NN + n];
  }
  Aall[((size_t)b * KP + c) * NN + n] = acc;
}

__global__ __launch_bounds__(256) void k_prepB(const float* __restrict__ u,
                                               const float* __restrict__ v,
                                               const int* __restrict__ ides,
                                               const float* __restrict__ Wgu,
                                               const float* __restrict__ Wgv,
                                               float* __restrict__ Ball) {
  int idx = blockIdx.x * 256 + threadIdx.x;
  if (idx >= BB * CH * SS) return;
  int s = idx % SS;
  int c = (idx / SS) % CH;
  int b = idx / (CH * SS);
  int gi = ides[b * SS + s];
  float acc = 0.f;
  if (c < 2 * CU) {
    for (int j = 0; j < CU; ++j)
      acc += Wgu[c * (2 * CU) + CU + j] * u[((size_t)b * CU + j) * NN + gi];
  } else {
    int o = c - 2 * CU;
    for (int j = 0; j < CV; ++j)
      acc += Wgv[o * (2 * CV) + CV + j] * v[((size_t)b * CV + j) * NN + gi];
  }
  Ball[((size_t)b * KP + c) * SS + s] = acc;
}

// ---------------- closed-form BN1/BN2 stats from row sums ----------------
__global__ __launch_bounds__(256) void k_rowstats(const float* __restrict__ Aall,
                                                  const float* __restrict__ Ball,
                                                  float* __restrict__ PA, float* __restrict__ PA2,
                                                  float* __restrict__ PB, float* __restrict__ PB2) {
  __shared__ float r1[256], r2[256];
  int b = blockIdx.x / CH, c = blockIdx.x % CH, tid = threadIdx.x;
  const float* row = &Aall[((size_t)b * KP + c) * NN];
  float s1 = 0.f, s2 = 0.f;
  for (int i = tid; i < NN; i += 256) { float x = row[i]; s1 += x; s2 += x * x; }
  r1[tid] = s1; r2[tid] = s2;
  __syncthreads();
  for (int off = 128; off > 0; off >>= 1) {
    if (tid < off) { r1[tid] += r1[tid + off]; r2[tid] += r2[tid + off]; }
    __syncthreads();
  }
  if (tid == 0) {
    PA[b * KP + c] = r1[0]; PA2[b * KP + c] = r2[0];
    const float* rb = &Ball[((size_t)b * KP + c) * SS];
    float t1 = 0.f, t2 = 0.f;
    for (int s = 0; s < SS; ++s) { float x = rb[s]; t1 += x; t2 += x * x; }
    PB[b * KP + c] = t1; PB2[b * KP + c] = t2;
  }
}

__global__ void k_scale12(const float* __restrict__ PA, const float* __restrict__ PA2,
                          const float* __restrict__ PB, const float* __restrict__ PB2,
                          const float* __restrict__ bgu, const float* __restrict__ g1, const float* __restrict__ b1,
                          const float* __restrict__ bgv, const float* __restrict__ g2, const float* __restrict__ b2,
                          float* __restrict__ cscale, float* __restrict__ cshift) {
  int c = threadIdx.x;
  if (c >= KP) return;
  if (c >= CH) { cscale[c] = 0.f; cshift[c] = 0.f; return; }
  float bias, gam, bet;
  if (c < 2 * CU) { bias = bgu[c]; gam = g1[c]; bet = b1[c]; }
  else            { bias = bgv[c - 2 * CU]; gam = g2[c - 2 * CU]; bet = b2[c - 2 * CU]; }
  float SA = 0.f, SA2 = 0.f, SB = 0.f, SB2 = 0.f, cr = 0.f;
  for (int b = 0; b < BB; ++b) {
    float pa = PA[b * KP + c], pb = PB[b * KP + c];
    SA += pa; SB += pb; SA2 += PA2[b * KP + c]; SB2 += PB2[b * KP + c]; cr += pa * pb;
  }
  float sumx  = (float)SS * SA + (float)NN * SB + CNT1 * bias;
  float sumx2 = (float)SS * SA2 + (float)NN * SB2 + 2.f * cr +
                2.f * bias * ((float)SS * SA + (float)NN * SB) + CNT1 * bias * bias;
  float mean = sumx / CNT1;
  float var  = sumx2 / CNT1 - mean * mean;
  float sc   = gam * rsqrtf(var + EPSBN);
  cscale[c] = sc;
  cshift[c] = bet - mean * sc;
}

// ---------------- fold BN affine into A/B terms; bf16 weight prep ----------------
__global__ __launch_bounds__(256) void k_applyAS(const float* __restrict__ Aall,
                                                 const float* __restrict__ cscale,
                                                 float* __restrict__ AS) {
  int idx = blockIdx.x * 256 + threadIdx.x;
  if (idx >= BB * KP * NN) return;
  int c = (idx / NN) % KP;
  AS[idx] = (c < CH) ? cscale[c] * Aall[idx] : 0.f;
}

__global__ __launch_bounds__(256) void k_applyBSW(const float* __restrict__ Ball,
                                                  const float* __restrict__ cscale,
                                                  const float* __restrict__ cshift,
                                                  const float* __restrict__ bgu,
                                                  const float* __restrict__ bgv,
                                                  const float* __restrict__ Wuv,
                                                  const float* __restrict__ Wf,
                                                  float* __restrict__ BS,
                                                  bf16_t* __restrict__ WuvB,
                                                  bf16_t* __restrict__ WfB) {
  int idx = blockIdx.x * 256 + threadIdx.x;
  const int NBS = BB * KP * SS;          // 20480
  const int NWU = CO * KP;               // 10240
  if (idx < NBS) {
    int c = (idx / SS) % KP;
    float v = 0.f;
    if (c < CH) {
      float bias = (c < 2 * CU) ? bgu[c] : bgv[c - 2 * CU];
      v = cscale[c] * (Ball[idx] + bias) + cshift[c];
    }
    BS[idx] = v;
  } else if (idx < NBS + NWU) {
    int j = idx - NBS;
    int k = j % KP, o = j / KP;
    WuvB[j] = (k < CH) ? (bf16_t)Wuv[o * CH + k] : (bf16_t)0.f;
  } else if (idx < NBS + NWU + CO * CO) {
    int j = idx - NBS - NWU;
    WfB[j] = (bf16_t)Wf[j];
  }
}

// ---------------- hot fused kernel: Wuv GEMM + BN3 stat accum + max/min over s ----------------
__global__ __launch_bounds__(256)
void k_rel_wmma(const float* __restrict__ AS, const float* __restrict__ BS,
                const bf16_t* __restrict__ WuvB, const float* __restrict__ buv,
                float* __restrict__ gmax, float* __restrict__ gmin,
                float* __restrict__ gsum, float* __restrict__ gsq) {
  __shared__ bf16_t sH[SS * KP];       // H^T tile: [s][c], relu'd separable values, bf16
  __shared__ bf16_t sW[CO * KP];       // Wuv padded: [o][k]
  __shared__ float  ytile[SS * YSTR];  // y staging: [s][o], padded stride
  __shared__ float  part[CO * 4 * 4];
  __shared__ float  aCol[KP];

  const int b = blockIdx.x >> 12;
  const int n = blockIdx.x & (NN - 1);
  const int tid = threadIdx.x;

  // async DMA of the (L2-resident) bf16 weight tile into LDS
  {
    const unsigned lds_base = (unsigned)(unsigned long long)&sW[0];
    #pragma unroll
    for (int j = 0; j < (CO * KP * 2) / (256 * 16); ++j) {   // 5 chunks of 16B per thread
      unsigned off = (tid + j * 256) * 16u;
      async_copy_b128(lds_base + off, off, WuvB);
    }
  }
  // stage the 160 A-column values once (instead of 10240 redundant global reads)
  if (tid < KP) aCol[tid] = AS[((size_t)b * KP + tid) * NN + n];
  __syncthreads();

  for (int idx = tid; idx < SS * KP; idx += 256) {
    int s = idx / KP, c = idx - s * KP;
    float h = aCol[c] + BS[((size_t)b * KP + c) * SS + s];
    sH[idx] = (bf16_t)fmaxf(h, 0.f);
  }
  async_wait();
  __syncthreads();

  const int wave = tid >> 5;
  const int lane = tid & 31;
  const int l16  = lane & 15;
  const int half = lane >> 4;

  #pragma unroll
  for (int it = 0; it < 2; ++it) {
    const int t  = wave * 2 + it;
    const int mo = (t >> 2) * 16;   // output-channel tile
    const int so = (t & 3) * 16;    // s tile
    v8f acc = {0.f, 0.f, 0.f, 0.f, 0.f, 0.f, 0.f, 0.f};
    #pragma unroll
    for (int kk = 0; kk < KP / 32; ++kk) {
      const int k0 = kk * 32;
      const v8bf* ap = (const v8bf*)&sW[(mo + l16) * KP + k0 + half * 8];
      const v8bf* bp = (const v8bf*)&sH[(so + l16) * KP + k0 + half * 8];
      v8bf alo = ap[0], ahi = ap[2];   // K+0..7 and K+16..23 (per-half offset)
      v8bf blo = bp[0], bhi = bp[2];
      v16bf af, bfm;
      #pragma unroll
      for (int i = 0; i < 8; ++i) { af[i] = alo[i]; af[i + 8] = ahi[i];
                                    bfm[i] = blo[i]; bfm[i + 8] = bhi[i]; }
      acc = __builtin_amdgcn_wmma_f32_16x16x32_bf16(false, af, false, bfm,
                                                    (short)0, acc, false, false);
    }
    // stage y (+buv) to LDS: lane covers row s=so+l16, 8 consecutive channels
    const float4* bv = (const float4*)&buv[mo + half * 8];
    float4 b0 = bv[0], b1 = bv[1];
    float4 y0, y1;
    y0.x = acc[0] + b0.x; y0.y = acc[1] + b0.y; y0.z = acc[2] + b0.z; y0.w = acc[3] + b0.w;
    y1.x = acc[4] + b1.x; y1.y = acc[5] + b1.y; y1.z = acc[6] + b1.z; y1.w = acc[7] + b1.w;
    float* yp = &ytile[(so + l16) * YSTR + mo + half * 8];
    *(float4*)yp       = y0;
    *(float4*)(yp + 4) = y1;
  }
  __syncthreads();

  // cooperative reduction over s: thread (o, quarter) scans 16 rows
  {
    const int o = tid & 63, q = tid >> 6;
    float vmax = -3.4e38f, vmin = 3.4e38f, vs = 0.f, vq2 = 0.f;
    #pragma unroll
    for (int i = 0; i < 16; ++i) {
      float y = ytile[(q * 16 + i) * YSTR + o];
      vmax = fmaxf(vmax, y); vmin = fminf(vmin, y); vs += y; vq2 += y * y;
    }
    float* pp = &part[(o * 4 + q) * 4];
    pp[0] = vmax; pp[1] = vmin; pp[2] = vs; pp[3] = vq2;
  }
  __syncthreads();
  if (tid < CO) {
    const int o = tid;
    float vmax = -3.4e38f, vmin = 3.4e38f, vs = 0.f, vq2 = 0.f;
    #pragma unroll
    for (int g = 0; g < 4; ++g) {
      const float* p = &part[(o * 4 + g) * 4];
      vmax = fmaxf(vmax, p[0]); vmin = fminf(vmin, p[1]); vs += p[2]; vq2 += p[3];
    }
    const size_t base = ((size_t)b * CO + o) * NN + n;
    gmax[base] = vmax; gmin[base] = vmin; gsum[base] = vs; gsq[base] = vq2;
  }
}

// ---------------- BN3 finalize; fused BN3+relu+max via max/min trick ----------------
__global__ __launch_bounds__(256) void k_stat3(const float* __restrict__ gsum,
                                               const float* __restrict__ gsq,
                                               float* __restrict__ sum3, float* __restrict__ sq3) {
  __shared__ float r1[256], r2[256];
  int o = blockIdx.x, tid = threadIdx.x;
  float s1 = 0.f, s2 = 0.f;
  for (int j = tid; j < BB * NN; j += 256) {
    int b = j >> 12, n = j & (NN - 1);
    size_t a = ((size_t)b * CO + o) * NN + n;
    s1 += gsum[a]; s2 += gsq[a];
  }
  r1[tid] = s1; r2[tid] = s2;
  __syncthreads();
  for (int off = 128; off > 0; off >>= 1) {
    if (tid < off) { r1[tid] += r1[tid + off]; r2[tid] += r2[tid + off]; }
    __syncthreads();
  }
  if (tid == 0) { sum3[o] = r1[0]; sq3[o] = r2[0]; }
}

__global__ void k_scale3(const float* __restrict__ sum3, const float* __restrict__ sq3,
                         const float* __restrict__ g3, const float* __restrict__ b3,
                         float* __restrict__ scale3, float* __restrict__ shift3) {
  int o = threadIdx.x;
  if (o >= CO) return;
  float mean = sum3[o] / CNT1;
  float var  = sq3[o] / CNT1 - mean * mean;
  float sc   = g3[o] * rsqrtf(var + EPSBN);
  scale3[o] = sc; shift3[o] = b3[o] - mean * sc;
}

__global__ __launch_bounds__(256) void k_hmax(const float* __restrict__ gmax,
                                              const float* __restrict__ gmin,
                                              const float* __restrict__ scale3,
                                              const float* __restrict__ shift3,
                                              bf16_t* __restrict__ hB) {
  int idx = blockIdx.x * 256 + threadIdx.x;
  if (idx >= BB * NN * CO) return;
  int o = idx & 63, n = (idx >> 6) & (NN - 1), b = idx >> 18;
  float sc = scale3[o], sh = shift3[o];
  size_t a = ((size_t)b * CO + o) * NN + n;
  float base = (sc >= 0.f) ? gmax[a] : gmin[a];   // monotone affine commutes with max over s
  hB[idx] = (bf16_t)fmaxf(sc * base + sh, 0.f);   // layout [b][n][o]
}

// ---------------- Wf GEMM via WMMA ----------------
__global__ __launch_bounds__(256)
void k_wf_wmma(const bf16_t* __restrict__ hB, const bf16_t* __restrict__ WfB,
               const float* __restrict__ bfv, float* __restrict__ z) {
  __shared__ bf16_t sH[64 * 64];   // [n][c]
  __shared__ bf16_t sW[64 * 64];   // [o][k]
  const int b  = blockIdx.x >> 6;
  const int n0 = (blockIdx.x & 63) * 64;
  const int tid = threadIdx.x;
  {
    const unsigned hbase = (unsigned)(unsigned long long)&sH[0];
    const unsigned wbase = (unsigned)(unsigned long long)&sW[0];
    const bf16_t* hsrc = &hB[((size_t)b * NN + n0) * 64];
    #pragma unroll
    for (int j = 0; j < (64 * 64 * 2) / (256 * 16); ++j) {   // 2 chunks of 16B per thread
      unsigned off = (tid + j * 256) * 16u;
      async_copy_b128(hbase + off, off, hsrc);
      async_copy_b128(wbase + off, off, WfB);
    }
  }
  async_wait();
  __syncthreads();
  const int wave = tid >> 5, lane = tid & 31, l16 = lane & 15, half = lane >> 4;
  #pragma unroll
  for (int it = 0; it < 2; ++it) {
    const int t = wave * 2 + it, mo = (t >> 2) * 16, so = (t & 3) * 16;
    v8f acc = {0.f, 0.f, 0.f, 0.f, 0.f, 0.f, 0.f, 0.f};
    #pragma unroll
    for (int kk = 0; kk < 2; ++kk) {
      const int k0 = kk * 32;
      const v8bf* ap = (const v8bf*)&sW[(mo + l16) * 64 + k0 + half * 8];
      const v8bf* bp = (const v8bf*)&sH[(so + l16) * 64 + k0 + half * 8];
      v8bf alo = ap[0], ahi = ap[2];
      v8bf blo = bp[0], bhi = bp[2];
      v16bf af, bfm;
      #pragma unroll
      for (int i = 0; i < 8; ++i) { af[i] = alo[i]; af[i + 8] = ahi[i];
                                    bfm[i] = blo[i]; bfm[i + 8] = bhi[i]; }
      acc = __builtin_amdgcn_wmma_f32_16x16x32_bf16(false, af, false, bfm,
                                                    (short)0, acc, false, false);
    }
    #pragma unroll
    for (int r = 0; r < 8; ++r) {
      const int o = mo + half * 8 + r;
      z[((size_t)b * CO + o) * NN + n0 + so + l16] = acc[r] + bfv[o];
    }
  }
}

// ---------------- BN4 + output assembly ----------------
__global__ __launch_bounds__(256) void k_stat4(const float* __restrict__ z,
                                               float* __restrict__ sum4, float* __restrict__ sq4) {
  __shared__ float r1[256], r2[256];
  int o = blockIdx.x, tid = threadIdx.x;
  float s1 = 0.f, s2 = 0.f;
  for (int j = tid; j < BB * NN; j += 256) {
    int b = j >> 12, n = j & (NN - 1);
    float x = z[((size_t)b * CO + o) * NN + n];
    s1 += x; s2 += x * x;
  }
  r1[tid] = s1; r2[tid] = s2;
  __syncthreads();
  for (int off = 128; off > 0; off >>= 1) {
    if (tid < off) { r1[tid] += r1[tid + off]; r2[tid] += r2[tid + off]; }
    __syncthreads();
  }
  if (tid == 0) { sum4[o] = r1[0]; sq4[o] = r2[0]; }
}

__global__ void k_scale4(const float* __restrict__ sum4, const float* __restrict__ sq4,
                         const float* __restrict__ g4, const float* __restrict__ b4,
                         float* __restrict__ scale4, float* __restrict__ shift4) {
  int o = threadIdx.x;
  if (o >= CO) return;
  float mean = sum4[o] / CNT2;
  float var  = sq4[o] / CNT2 - mean * mean;
  float sc   = g4[o] * rsqrtf(var + EPSBN);
  scale4[o] = sc; shift4[o] = b4[o] - mean * sc;
}

__global__ __launch_bounds__(256) void k_out(const float* __restrict__ u,
                                             const float* __restrict__ v,
                                             const float* __restrict__ z,
                                             const float* __restrict__ scale4,
                                             const float* __restrict__ shift4,
                                             float* __restrict__ out) {
  int idx = blockIdx.x * 256 + threadIdx.x;
  if (idx >= BB * (CV + CU) * NN) return;
  int n  = idx % NN;
  int ch = (idx / NN) % (CV + CU);
  int b  = idx / ((CV + CU) * NN);
  float val;
  if (ch < CV) {
    float zz = z[((size_t)b * CO + ch) * NN + n];
    val = v[((size_t)b * CV + ch) * NN + n] + fmaxf(scale4[ch] * zz + shift4[ch], 0.f);
  } else {
    val = u[((size_t)b * CU + (ch - CV)) * NN + n];
  }
  out[idx] = val;
}

// ---------------- host launcher ----------------
extern "C" void kernel_launch(void* const* d_in, const int* in_sizes, int n_in,
                              void* d_out, int out_size, void* d_ws, size_t ws_size,
                              hipStream_t stream) {
  const float* u   = (const float*)d_in[0];
  const float* v   = (const float*)d_in[1];
  const int*   ide = (const int*)  d_in[2];
  const float* Wgu = (const float*)d_in[3];
  const float* bgu = (const float*)d_in[4];
  const float* g1  = (const float*)d_in[5];
  const float* b1  = (const float*)d_in[6];
  const float* Wgv = (const float*)d_in[7];
  const float* bgv = (const float*)d_in[8];
  const float* g2  = (const float*)d_in[9];
  const float* b2  = (const float*)d_in[10];
  const float* Wuv = (const float*)d_in[11];
  const float* buv = (const float*)d_in[12];
  const float* g3  = (const float*)d_in[13];
  const float* b3  = (const float*)d_in[14];
  const float* Wf  = (const float*)d_in[15];
  const float* bfv = (const float*)d_in[16];
  const float* g4  = (const float*)d_in[17];
  const float* b4  = (const float*)d_in[18];
  float* out = (float*)d_out;

  char* p = (char*)d_ws;
  auto carve = [&p](size_t bytes) -> char* {
    char* r = p;
    p += (bytes + 255) & ~(size_t)255;
    return r;
  };
  float*  Aall = (float*)carve((size_t)BB * KP * NN * 4);
  float*  Ball = (float*)carve((size_t)BB * KP * SS * 4);
  float*  PA   = (float*)carve(BB * KP * 4);
  float*  PA2  = (float*)carve(BB * KP * 4);
  float*  PB   = (float*)carve(BB * KP * 4);
  float*  PB2  = (float*)carve(BB * KP * 4);
  float*  csc  = (float*)carve(KP * 4);
  float*  csh  = (float*)carve(KP * 4);
  float*  AS   = (float*)carve((size_t)BB * KP * NN * 4);
  float*  BS   = (float*)carve((size_t)BB * KP * SS * 4);
  bf16_t* WuvB = (bf16_t*)carve((size_t)CO * KP * 2);
  bf16_t* WfB  = (bf16_t*)carve((size_t)CO * CO * 2);
  float*  gmax = (float*)carve((size_t)BB * CO * NN * 4);
  float*  gmin = (float*)carve((size_t)BB * CO * NN * 4);
  float*  gsum = (float*)carve((size_t)BB * CO * NN * 4);
  float*  gsq  = (float*)carve((size_t)BB * CO * NN * 4);
  float*  sum3 = (float*)carve(CO * 4);
  float*  sq3  = (float*)carve(CO * 4);
  float*  sc3  = (float*)carve(CO * 4);
  float*  sh3  = (float*)carve(CO * 4);
  bf16_t* hB   = (bf16_t*)carve((size_t)BB * NN * CO * 2);
  float*  z    = (float*)carve((size_t)BB * CO * NN * 4);
  float*  sum4 = (float*)carve(CO * 4);
  float*  sq4  = (float*)carve(CO * 4);
  float*  sc4  = (float*)carve(CO * 4);
  float*  sh4  = (float*)carve(CO * 4);
  (void)ws_size; (void)in_sizes; (void)n_in; (void)out_size;

  k_prepA<<<(BB * CH * NN + 255) / 256, 256, 0, stream>>>(u, v, Wgu, Wgv, Aall);
  k_prepB<<<(BB * CH * SS + 255) / 256, 256, 0, stream>>>(u, v, ide, Wgu, Wgv, Ball);
  k_rowstats<<<BB * CH, 256, 0, stream>>>(Aall, Ball, PA, PA2, PB, PB2);
  k_scale12<<<1, 256, 0, stream>>>(PA, PA2, PB, PB2, bgu, g1, b1, bgv, g2, b2, csc, csh);
  k_applyAS<<<(BB * KP * NN + 255) / 256, 256, 0, stream>>>(Aall, csc, AS);
  k_applyBSW<<<(BB * KP * SS + CO * KP + CO * CO + 255) / 256, 256, 0, stream>>>(
      Ball, csc, csh, bgu, bgv, Wuv, Wf, BS, WuvB, WfB);
  k_rel_wmma<<<BB * NN, 256, 0, stream>>>(AS, BS, WuvB, buv, gmax, gmin, gsum, gsq);
  k_stat3<<<CO, 256, 0, stream>>>(gsum, gsq, sum3, sq3);
  k_scale3<<<1, 64, 0, stream>>>(sum3, sq3, g3, b3, sc3, sh3);
  k_hmax<<<(BB * NN * CO + 255) / 256, 256, 0, stream>>>(gmax, gmin, sc3, sh3, hB);
  k_wf_wmma<<<BB * 64, 256, 0, stream>>>(hB, WfB, bfv, z);
  k_stat4<<<CO, 256, 0, stream>>>(z, sum4, sq4);
  k_scale4<<<1, 64, 0, stream>>>(sum4, sq4, g4, b4, sc4, sh4);
  k_out<<<(BB * (CV + CU) * NN + 255) / 256, 256, 0, stream>>>(u, v, z, sc4, sh4, out);
}
